// TFCM_6322191859857
// MI455X (gfx1250) — matrix operand order repaired and checked by reading, not compiled, not run on hardware
//
#include <hip/hip_runtime.h>

// TFCM: temporal cross-attention fusion.
//   inputs: feats[8,384,40,40] f32, refs[4,8,384,40,40] f32, sopm[8,1,160,160] f32,
//           log_tau[1] f32, alpha_raw[1] f32
//   output: enhanced[8,384,40,40] f32
// Flash-attention over 4 ref frames with v_wmma_f32_16x16x32_f16, f32 accumulation.
// Q fragments are register-resident (constant across refs/keys); K/V staged in LDS.
// LDS = 151.8 KB -> 2 workgroups per WGP.

#define BB 8
#define CCH 384          // channels
#define HH 40
#define NN 1600          // tokens
#define NREF 4
#define CP 192           // CCH/2 : f16 pairs along C
#define MT 64            // query rows per workgroup (4 waves x 16)
#define KT 32            // keys per inner step
#define NWAVE 4

typedef _Float16 v16h __attribute__((ext_vector_type(16)));
typedef float    v8f  __attribute__((ext_vector_type(8)));
typedef unsigned int v4u __attribute__((ext_vector_type(4)));

union Frag { v4u u[2]; v16h h; };

__device__ __forceinline__ unsigned pk2(float a, float b) {
  union { _Float16 h[2]; unsigned u; } x;
  x.h[0] = (_Float16)a; x.h[1] = (_Float16)b;
  return x.u;
}

// ---------------- kernel 1: inverse L2 norms (cur + 4 refs) ----------------
__global__ void tfcm_norms(const float* __restrict__ feats,
                           const float* __restrict__ refs,
                           float* __restrict__ invn) {
  const int BN = BB * NN;
  int t = blockIdx.x * blockDim.x + threadIdx.x;
  if (t >= 5 * BN) return;
  int which = t / BN;                 // 0 = cur, 1..4 = ref r-1
  int bn = t - which * BN;
  int b = bn / NN, n = bn - b * NN;
  const float* src = (which == 0) ? feats
                                  : (refs + (size_t)(which - 1) * BB * CCH * NN);
  const float* p = src + (size_t)b * CCH * NN + n;
  float s = 0.f;
  for (int c = 0; c < CCH; ++c) { float v = p[(size_t)c * NN]; s += v * v; }
  invn[t] = 1.0f / fmaxf(sqrtf(s), 1e-12f);
}

// ---------------- kernel 2: fused flash attention + blend ------------------
// LDS layout (bytes):
//   Khp  [KT][CP]    u32 @ 0       (24576)
//   Vhp  [CCH][16]   u32 @ 24576   (24576)
//   Php  [4][16][16] u32 @ 49152   (4096)
//   tot  [4][16][CCH]f32 @ 53248   (98304)
//   mskv [MT]        f32 @ 151552  (256)
#define SMEM_BYTES 151808

__global__ __launch_bounds__(128, 1)
void tfcm_attn(const float* __restrict__ feats,
               const float* __restrict__ refs,
               const float* __restrict__ sopm,
               const float* __restrict__ log_tau,
               const float* __restrict__ alpha_raw,
               const float* __restrict__ invn,
               float* __restrict__ out) {
  extern __shared__ char smem[];
  unsigned* Khp = (unsigned*)(smem);
  unsigned* Vhp = (unsigned*)(smem + 24576);
  unsigned* Php = (unsigned*)(smem + 49152);
  float*    tot = (float*)(smem + 53248);
  float*    mskv = (float*)(smem + 151552);

  const int n0   = blockIdx.x * MT;
  const int b    = blockIdx.y;
  const int tid  = threadIdx.x;
  const int wave = tid >> 5;
  const int lane = tid & 31;
  const int ln   = lane & 15;
  const int h    = lane >> 4;

  // scalar params
  const float L2E = 1.4426950408889634f;
  float tau = __builtin_amdgcn_exp2f(log_tau[0] * L2E);
  tau = fminf(fmaxf(tau, 0.005f), 0.1f);
  const float sscale = L2E / tau;                    // logits in log2 domain
  const float alpha = 1.0f / (1.0f + __builtin_amdgcn_exp2f(-alpha_raw[0] * L2E));

  // ---- Q tile: 12 register-resident A fragments (constant over refs/keys) ----
  // A-frag element: row q = ln, VGPR j<4: c = 32f+2j+8h ; j>=4: c = 32f+16+2(j-4)+8h
  Frag qf[12];
  {
    const int q = n0 + wave * 16 + ln;
    const float iq = invn[b * NN + q];
    const float* qb = feats + (size_t)b * CCH * NN + q;
    #pragma unroll
    for (int f = 0; f < 12; ++f) {
      #pragma unroll
      for (int j = 0; j < 4; ++j) {
        int c0 = 32 * f + 2 * j + 8 * h;
        int c1 = c0 + 16;
        qf[f].u[0][j] = pk2(qb[(size_t)c0 * NN] * iq, qb[(size_t)(c0 + 1) * NN] * iq);
        qf[f].u[1][j] = pk2(qb[(size_t)c1 * NN] * iq, qb[(size_t)(c1 + 1) * NN] * iq);
      }
    }
  }

  // ---- SOPM bilinear 160->40 (align_corners=False) + threshold ----
  if (tid < MT) {
    int n = n0 + tid;
    int yi = n / HH, xj = n - yi * HH;
    int y1 = 4 * yi + 1, x1 = 4 * xj + 1;
    const float* sp = sopm + (size_t)b * 160 * 160;
    float v = 0.25f * (sp[y1 * 160 + x1]       + sp[y1 * 160 + x1 + 1] +
                       sp[(y1 + 1) * 160 + x1] + sp[(y1 + 1) * 160 + x1 + 1]);
    mskv[tid] = (v > 0.3f) ? 1.0f : 0.0f;
  }

  const v8f vzero = {0.f,0.f,0.f,0.f,0.f,0.f,0.f,0.f};

  for (int r = 0; r < NREF; ++r) {
    const float* rbase = refs + ((size_t)r * BB + b) * CCH * NN;
    const float* rinv  = invn + (1 + r) * BB * NN + b * NN;

    v8f acc[24];
    #pragma unroll
    for (int ch = 0; ch < 24; ++ch) acc[ch] = vzero;
    float l[8];
    #pragma unroll
    for (int g = 0; g < 8; ++g) l[g] = 0.0f;

    for (int k0 = 0; k0 < NN; k0 += KT) {
      __syncthreads();                       // prior step done reading K/V
      // ---- stage K (normalized) and V (raw) tiles, f16 pairs ----
      float ik = rinv[k0 + lane];
      for (int cc = wave; cc < CP; cc += NWAVE) {
        int c0 = cc * 2;
        float f0 = rbase[(size_t)c0 * NN + k0 + lane];
        float f1 = rbase[(size_t)(c0 + 1) * NN + k0 + lane];
        Khp[lane * CP + cc] = pk2(f0 * ik, f1 * ik);      // [key][cc]
        float g0 = __shfl_xor(f0, 1, 32);
        float g1 = __shfl_xor(f1, 1, 32);
        if (!(lane & 1)) {                                 // pack key pairs
          int kk = lane >> 1;
          Vhp[c0 * 16 + kk]       = pk2(f0, g0);           // [c][kk]
          Vhp[(c0 + 1) * 16 + kk] = pk2(f1, g1);
        }
      }
      __syncthreads();

      // ---- S(16x32) = Q(16xC) * K^T(Cx32) : 24 WMMAs over C ----
      v8f S0 = vzero, S1 = vzero;
      const unsigned* K0 = Khp + ln * CP;
      const unsigned* K1 = Khp + (ln + 16) * CP;
      #pragma unroll
      for (int f = 0; f < 12; ++f) {
        int cb = f * 16;
        Frag b0, b1;
        b0.u[0] = *(const v4u*)(K0 + cb + 4 * h);
        b0.u[1] = *(const v4u*)(K0 + cb + 8 + 4 * h);
        b1.u[0] = *(const v4u*)(K1 + cb + 4 * h);
        b1.u[1] = *(const v4u*)(K1 + cb + 8 + 4 * h);
        S0 = __builtin_amdgcn_wmma_f32_16x16x32_f16(false, qf[f].h, false, b0.h,
                                                    (short)0, S0, false, false);
        S1 = __builtin_amdgcn_wmma_f32_16x16x32_f16(false, qf[f].h, false, b1.h,
                                                    (short)0, S1, false, false);
      }

      // ---- softmax weights: cosine logits bounded by 1/tau, fixed shift ----
      unsigned* Pw = Php + wave * 256;
      #pragma unroll
      for (int g = 0; g < 8; ++g) {
        float e0 = __builtin_amdgcn_exp2f(S0[g] * sscale - sscale);
        float e1 = __builtin_amdgcn_exp2f(S1[g] * sscale - sscale);
        float rs = e0 + e1;
        rs += __shfl_xor(rs, 1, 32);
        rs += __shfl_xor(rs, 2, 32);
        rs += __shfl_xor(rs, 4, 32);
        rs += __shfl_xor(rs, 8, 32);        // per-row sum (halves independent)
        l[g] += rs;
        float p0 = __shfl_xor(e0, 1, 32);
        float p1 = __shfl_xor(e1, 1, 32);
        if (!(lane & 1)) {
          int row = g + 8 * h;
          int col = ln >> 1;
          Pw[row * 16 + col]     = pk2(e0, p0);
          Pw[row * 16 + 8 + col] = pk2(e1, p1);
        }
      }
      asm volatile("s_wait_dscnt 0" ::: "memory");

      // ---- acc(16x384) += P(16x32) * V(32x384) : 24 WMMAs ----
      Frag ap;
      const unsigned* Pq = Php + wave * 256 + ln * 16;
      ap.u[0] = *(const v4u*)(Pq + 8 * h);
      ap.u[1] = *(const v4u*)(Pq + 8 * h + 4);
      #pragma unroll
      for (int ch = 0; ch < 24; ++ch) {
        Frag bv;
        const unsigned* Vb = Vhp + (ch * 16 + ln) * 16 + 8 * h;
        bv.u[0] = *(const v4u*)(Vb);
        bv.u[1] = *(const v4u*)(Vb + 4);
        acc[ch] = __builtin_amdgcn_wmma_f32_16x16x32_f16(false, ap.h, false, bv.h,
                                                         (short)0, acc[ch],
                                                         false, false);
      }
    }

    // ---- fold this ref into the running total with per-row 1/l ----
    float rl[8];
    #pragma unroll
    for (int g = 0; g < 8; ++g) rl[g] = __builtin_amdgcn_rcpf(fmaxf(l[g], 1e-30f));
    float* tw = tot + wave * 16 * CCH;
    #pragma unroll
    for (int ch = 0; ch < 24; ++ch) {
      #pragma unroll
      for (int g = 0; g < 8; ++g) {
        int q = g + 8 * h;
        int c = ch * 16 + ln;
        float v = acc[ch][g] * rl[g];
        if (r == 0) tw[q * CCH + c] = v;
        else        tw[q * CCH + c] += v;
      }
    }
  }
  __syncthreads();

  // ---- epilogue: warped/4, alpha blend, SOPM mask, write [B,C,H,W] ----
  for (int i = tid; i < MT * CCH; i += 128) {
    int q = i & (MT - 1);
    int c = i >> 6;
    int n = n0 + q;
    size_t gi = ((size_t)b * CCH + c) * NN + n;
    float t = feats[gi];
    float w = tot[(q >> 4) * 16 * CCH + (q & 15) * CCH + c] * 0.25f;
    float fused = alpha * w + (1.0f - alpha) * t;
    float m = mskv[q];
    out[gi] = m * fused + (1.0f - m) * t;
  }
}

extern "C" void kernel_launch(void* const* d_in, const int* in_sizes, int n_in,
                              void* d_out, int out_size, void* d_ws, size_t ws_size,
                              hipStream_t stream) {
  const float* feats     = (const float*)d_in[0];
  const float* refs      = (const float*)d_in[1];
  const float* sopm      = (const float*)d_in[2];
  const float* log_tau   = (const float*)d_in[3];
  const float* alpha_raw = (const float*)d_in[4];
  float* invn = (float*)d_ws;           // 5*B*N floats = 256 KB
  float* out  = (float*)d_out;

  int ntok = 5 * BB * NN;
  tfcm_norms<<<(ntok + 255) / 256, 256, 0, stream>>>(feats, refs, invn);

  (void)hipFuncSetAttribute(reinterpret_cast<const void*>(tfcm_attn),
                            hipFuncAttributeMaxDynamicSharedMemorySize,
                            SMEM_BYTES);
  dim3 grid(NN / MT, BB);
  tfcm_attn<<<grid, 128, SMEM_BYTES, stream>>>(feats, refs, sopm, log_tau,
                                               alpha_raw, invn, out);
}
